// LlamaAttention_8504035246558
// MI455X (gfx1250) — matrix operand order repaired and checked by reading, not compiled, not run on hardware
//
#include <hip/hip_runtime.h>
#include <hip/hip_bf16.h>
#include <math.h>

// ---------------------------------------------------------------------------
// MI455X (gfx1250, wave32) quantized Llama attention.
// GEMMs: LDS-tiled WMMA (64x128 workgroup tile, 8 waves, 32x32 C per wave)
// with double-buffered async global->LDS staging (ASYNCcnt path) when the
// toolchain exposes the gfx1250 async builtins.
// ---------------------------------------------------------------------------

typedef __attribute__((ext_vector_type(16))) _Float16 v16h;
typedef __attribute__((ext_vector_type(8)))  float    v8f;
typedef int gvec4i __attribute__((vector_size(16))); // matches builtin proto

#define AS1 __attribute__((address_space(1)))
#define AS3 __attribute__((address_space(3)))

#if defined(__has_builtin)
#if __has_builtin(__builtin_amdgcn_global_load_async_to_lds_b128) && \
    __has_builtin(__builtin_amdgcn_s_wait_asynccnt)
#define USE_ASYNC_LDS 1
#endif
#endif
#ifndef USE_ASYNC_LDS
#define USE_ASYNC_LDS 0
#endif

namespace {
constexpr int B_   = 2;
constexpr int S_   = 1024;
constexpr int HID  = 4096;
constexpr int NH   = 32;
constexpr int NKV  = 8;
constexpr int HD   = 128;
constexpr int GROUPS = NH / NKV;           // 4
constexpr int BT   = B_ * S_;              // 2048 token rows
constexpr float SCALING = 0.08838834764831845f; // HD^-0.5
constexpr float QMAX = 255.0f;             // 8-bit asymmetric

// GEMM tiling
constexpr int MT = 64;     // workgroup M tile
constexpr int NT = 128;    // workgroup N tile
constexpr int KC = 32;     // K slab per stage (one WMMA K)
constexpr int APITCH = 40; // halfs per LDS row (32 data + 8 pad -> 80B pitch,
                           // 20-bank stride => conflict-free b128 reads)
}

// ------------------------- async / sync LDS staging -------------------------
__device__ __forceinline__ void copy16_to_lds(const _Float16* g, _Float16* l) {
#if USE_ASYNC_LDS
  __builtin_amdgcn_global_load_async_to_lds_b128(
      (AS1 gvec4i*)g, (AS3 gvec4i*)l, 0, 0);
#else
  *(uint4*)l = *(const uint4*)g;
#endif
}
__device__ __forceinline__ void wait_lds_stage() {
#if USE_ASYNC_LDS
  __builtin_amdgcn_s_wait_asynccnt(0);
#endif
}

// ----------------------------- wave reductions -----------------------------
__device__ __forceinline__ float wave_max(float v) {
  #pragma unroll
  for (int o = 16; o > 0; o >>= 1) v = fmaxf(v, __shfl_xor(v, o, 32));
  return v;
}
__device__ __forceinline__ float wave_min(float v) {
  #pragma unroll
  for (int o = 16; o > 0; o >>= 1) v = fminf(v, __shfl_xor(v, o, 32));
  return v;
}
__device__ __forceinline__ float wave_sum(float v) {
  #pragma unroll
  for (int o = 16; o > 0; o >>= 1) v += __shfl_xor(v, o, 32);
  return v;
}

// ----------------------------- fp32 -> fp16 --------------------------------
__global__ void f32_to_f16_kernel(const float* __restrict__ in,
                                  _Float16* __restrict__ out, size_t n) {
  size_t i = (size_t)blockIdx.x * blockDim.x + threadIdx.x;
  size_t stride = (size_t)gridDim.x * blockDim.x;
  for (; i < n; i += stride) out[i] = (_Float16)in[i];
}

// ------------------- C[M,N] = A[M,K] * Bw[N,K]^T  (WMMA) --------------------
// 256 threads = 8 waves in a 2(M) x 4(N) grid; each wave owns a 32x32 C block
// (2x2 f32 accumulators). A (64xKC) and Bw (128xKC) slabs are double-buffered
// in LDS via async global->LDS b128 copies. Fragment layouts per ISA 7.12.2.
__global__ void gemm_xwT_f16_tiled(const _Float16* __restrict__ A,
                                   const _Float16* __restrict__ Bw,
                                   float* __restrict__ C,
                                   int M, int N, int K) {
  __shared__ _Float16 lA[2][MT][APITCH];
  __shared__ _Float16 lB[2][NT][APITCH];
  int tid = threadIdx.x;
  int nbx = N / NT;
  int bx = blockIdx.x % nbx;
  int by = blockIdx.x / nbx;
  int m0 = by * MT, n0 = bx * NT;
  (void)M;
  int wave = tid >> 5, lane = tid & 31;
  int wy = wave >> 2, wx = wave & 3;          // 2 x 4 wave grid
  int half = lane >> 4, col = lane & 15;

  // staging assignment: A = 256 b128 chunks (1/thread), B = 512 (2/thread)
  int arow = tid >> 2, ach = (tid & 3) * 8;
  int b0 = tid, b1 = tid + 256;

  const _Float16* gA  = A  + (size_t)(m0 + arow) * K + ach;
  const _Float16* gB0 = Bw + (size_t)(n0 + (b0 >> 2)) * K + (b0 & 3) * 8;
  const _Float16* gB1 = Bw + (size_t)(n0 + (b1 >> 2)) * K + (b1 & 3) * 8;

  auto stage = [&](int buf, int kc) {
    __builtin_prefetch(gA + kc + 2 * KC, 0, 1);   // L2 lookahead
    __builtin_prefetch(gB0 + kc + 2 * KC, 0, 1);
    copy16_to_lds(gA + kc,  &lA[buf][arow][ach]);
    copy16_to_lds(gB0 + kc, &lB[buf][b0 >> 2][(b0 & 3) * 8]);
    copy16_to_lds(gB1 + kc, &lB[buf][b1 >> 2][(b1 & 3) * 8]);
  };

  v8f acc[2][2] = {};
  int nk = K / KC;
  stage(0, 0);
  for (int kc = 0; kc < nk; ++kc) {
    wait_lds_stage();        // this wave's async copies into buf done
    __syncthreads();         // everyone's copies visible / prev reads done
    if (kc + 1 < nk) stage((kc + 1) & 1, (kc + 1) * KC);
    int buf = kc & 1;
    v16h af[2], bf[2];
    #pragma unroll
    for (int s = 0; s < 2; ++s) {
      const _Float16* ar = &lA[buf][wy * 32 + s * 16 + col][0];
      #pragma unroll
      for (int i = 0; i < 8; ++i) {
        af[s][i]     = ar[half * 8 + i];
        af[s][8 + i] = ar[16 + half * 8 + i];
      }
      const _Float16* br = &lB[buf][wx * 32 + s * 16 + col][0];
      #pragma unroll
      for (int j = 0; j < 16; ++j) bf[s][j] = br[(j >> 3) * 16 + half * 8 + (j & 7)];
    }
    #pragma unroll
    for (int i = 0; i < 2; ++i)
      #pragma unroll
      for (int j = 0; j < 2; ++j)
        acc[i][j] = __builtin_amdgcn_wmma_f32_16x16x32_f16(
            false, af[i], false, bf[j], (short)0, acc[i][j], false, false);
  }
  #pragma unroll
  for (int i = 0; i < 2; ++i) {
    int mrow = m0 + wy * 32 + i * 16 + half * 8;
    #pragma unroll
    for (int j = 0; j < 2; ++j) {
      float* cp = C + (size_t)mrow * N + n0 + wx * 32 + j * 16 + col;
      #pragma unroll
      for (int r = 0; r < 8; ++r) cp[(size_t)r * N] = acc[i][j][r];
    }
  }
}

// ------------------------------- RoPE (in place) ----------------------------
__global__ void rope_kernel(float* __restrict__ X,
                            const float* __restrict__ cosp,
                            const float* __restrict__ sinp, int H) {
  int total = BT * H * (HD / 2);
  int tid = blockIdx.x * blockDim.x + threadIdx.x;
  if (tid >= total) return;
  int dh = tid & 63;
  int h  = (tid >> 6) % H;
  int bs = tid / (64 * H);
  float* p = X + (size_t)bs * H * HD + h * HD;
  float q1 = p[dh], q2 = p[dh + 64];
  float c1 = cosp[(size_t)bs * HD + dh],      s1 = sinp[(size_t)bs * HD + dh];
  float c2 = cosp[(size_t)bs * HD + dh + 64], s2 = sinp[(size_t)bs * HD + dh + 64];
  p[dh]      = q1 * c1 - q2 * s1;
  p[dh + 64] = q2 * c2 + q1 * s2;
}

// --------------------- mean of K over sequence (per b,h,d) ------------------
__global__ void kmean_kernel(const float* __restrict__ Kf,
                             float* __restrict__ kmean) {
  int b = blockIdx.x / NKV, h = blockIdx.x % NKV, d = threadIdx.x;
  float s = 0.f;
  for (int t = 0; t < S_; ++t)
    s += Kf[(size_t)(b * S_ + t) * (NKV * HD) + h * HD + d];
  kmean[(size_t)(b * NKV + h) * HD + d] = s * (1.0f / S_);
}

// ------------- per-row (last-dim, len 128) asymmetric fake-quant ------------
__global__ void quant_rows_kernel(const float* __restrict__ in,
                                  const float* __restrict__ mean, // nullable
                                  _Float16* __restrict__ out,
                                  int nrows, int hcount) {
  int wid  = (int)(((size_t)blockIdx.x * blockDim.x + threadIdx.x) >> 5);
  int lane = threadIdx.x & 31;
  if (wid >= nrows) return;
  int h = wid % hcount;
  int b = wid / (S_ * hcount);
  const float* row = in + (size_t)wid * HD;
  float vals[4], mn = 3.4e38f, mx = -3.4e38f;
  #pragma unroll
  for (int j = 0; j < 4; ++j) {
    int e = lane + 32 * j;
    float v = row[e];
    if (mean) v -= mean[(size_t)(b * hcount + h) * HD + e];
    vals[j] = v;
    mn = fminf(mn, v);
    mx = fmaxf(mx, v);
  }
  mn = fminf(wave_min(mn), 0.f);
  mx = fmaxf(wave_max(mx), 0.f);
  float rng   = mx - mn;
  float scale = (rng == 0.f) ? 1.f : rng / QMAX;
  float zero  = rintf(-mn / scale);
  #pragma unroll
  for (int j = 0; j < 4; ++j) {
    int e = lane + 32 * j;
    float q = fminf(fmaxf(rintf(vals[j] / scale) + zero, 0.f), QMAX);
    out[(size_t)wid * HD + e] = (_Float16)((q - zero) * scale);
  }
}

// ----- V: fake-quant per (b,h,d) over seq; store transposed [b,h,d,s] f16 ---
__global__ void vquant_kernel(const float* __restrict__ Vf,
                              _Float16* __restrict__ Vtq) {
  int b = blockIdx.x / NKV, h = blockIdx.x % NKV, d = threadIdx.x;
  const float* src = Vf + (size_t)b * S_ * (NKV * HD) + h * HD + d;
  float mn = 3.4e38f, mx = -3.4e38f;
  for (int t = 0; t < S_; ++t) {
    float v = src[(size_t)t * (NKV * HD)];
    mn = fminf(mn, v); mx = fmaxf(mx, v);
  }
  mn = fminf(mn, 0.f); mx = fmaxf(mx, 0.f);
  float rng   = mx - mn;
  float scale = (rng == 0.f) ? 1.f : rng / QMAX;
  float zero  = rintf(-mn / scale);
  _Float16* dst = Vtq + (size_t)((b * NKV + h) * HD + d) * S_;
  for (int t = 0; t < S_; ++t) {
    float v = src[(size_t)t * (NKV * HD)];
    float q = fminf(fmaxf(rintf(v / scale) + zero, 0.f), QMAX);
    dst[t] = (_Float16)((q - zero) * scale);
  }
}

// ------------------------------ attention core ------------------------------
// One wave per (b, h, 16-query tile). Scores via WMMA, causal-masked softmax,
// w fake-quant (fp32 to d_out), then attn = w @ V^T via WMMA with the K loop
// bounded by the causal frontier. LDS: 16x1024 f32 scores + 16x1024 f16 w.
__global__ void attention_kernel(const _Float16* __restrict__ Qq,
                                 const _Float16* __restrict__ Kq,
                                 const _Float16* __restrict__ Vtq,
                                 float* __restrict__ Wout,
                                 _Float16* __restrict__ AttnH) {
  extern __shared__ char smem[];
  float*    s_s = (float*)smem;                        // 64 KB
  _Float16* s_w = (_Float16*)(smem + 16 * S_ * 4);     // 32 KB
  int blk = blockIdx.x;
  int qt = blk % (S_ / 16);
  int h  = (blk / (S_ / 16)) % NH;
  int b  = blk / ((S_ / 16) * NH);
  int hk = h / GROUPS;
  int lane = threadIdx.x & 31;
  int half = lane >> 4;
  int col  = lane & 15;
  int q0 = qt * 16;

  const _Float16* qrow = Qq + (size_t)(b * S_ + q0 + col) * HID + h * HD;
  v16h aq[4];
  #pragma unroll
  for (int c = 0; c < 4; ++c) {
    #pragma unroll
    for (int i = 0; i < 8; ++i) {
      aq[c][i]     = qrow[c * 32 + half * 8 + i];
      aq[c][8 + i] = qrow[c * 32 + 16 + half * 8 + i];
    }
  }

  // ---- scores = q @ k^T (scaled, causal) into LDS ----
  for (int n0 = 0; n0 < S_; n0 += 16) {
    if (n0 <= q0 + 15) {
      const _Float16* krow = Kq + (size_t)(b * S_ + n0 + col) * (NKV * HD) + hk * HD;
      v8f acc = {};
      #pragma unroll
      for (int c = 0; c < 4; ++c) {
        v16h bk;
        #pragma unroll
        for (int j = 0; j < 16; ++j) bk[j] = krow[c * 32 + (j >> 3) * 16 + half * 8 + (j & 7)];
        acc = __builtin_amdgcn_wmma_f32_16x16x32_f16(false, aq[c], false, bk,
                                                     (short)0, acc, false, false);
      }
      #pragma unroll
      for (int r = 0; r < 8; ++r) {
        int row = half * 8 + r, cc = n0 + col;
        float m = (cc <= q0 + row) ? 0.0f : -1.0e9f;
        s_s[row * S_ + cc] = acc[r] * SCALING + m;
      }
    } else {
      #pragma unroll
      for (int r = 0; r < 8; ++r)
        s_s[(half * 8 + r) * S_ + n0 + col] = -1.0e9f;
    }
  }

  // ---- per-row softmax + 8-bit fake-quant of w ----
  for (int row = 0; row < 16; ++row) {
    float* sr = s_s + row * S_;
    float mx = -3.4e38f;
    for (int t = lane; t < S_; t += 32) mx = fmaxf(mx, sr[t]);
    mx = wave_max(mx);
    float sum = 0.f;
    for (int t = lane; t < S_; t += 32) {
      float e = __expf(sr[t] - mx);
      sr[t] = e; sum += e;
    }
    sum = wave_sum(sum);
    float inv = 1.0f / sum;
    float wmn = 0.f, wmx = 0.f;
    for (int t = lane; t < S_; t += 32) {
      float w = sr[t] * inv;
      sr[t] = w;
      wmn = fminf(wmn, w); wmx = fmaxf(wmx, w);
    }
    wmn = fminf(wave_min(wmn), 0.f);
    wmx = fmaxf(wave_max(wmx), 0.f);
    float rng   = wmx - wmn;
    float scale = (rng == 0.f) ? 1.f : rng / QMAX;
    float zero  = rintf(-wmn / scale);
    size_t wbase = (((size_t)(b * NH + h)) * S_ + (q0 + row)) * (size_t)S_;
    for (int t = lane; t < S_; t += 32) {
      float q  = fminf(fmaxf(rintf(sr[t] / scale) + zero, 0.f), QMAX);
      float dq = (q - zero) * scale;
      Wout[wbase + t]  = dq;             // 2nd output (fp32)
      s_w[row * S_ + t] = (_Float16)dq;  // GEMM operand
    }
  }

  // ---- attn = w @ V^T (only the un-masked K range contributes) ----
  int kend = q0 + 16;
  if (kend & 31) kend += 16;
  for (int dt = 0; dt < HD / 16; ++dt) {
    const _Float16* vrow = Vtq + (size_t)((b * NKV + hk) * HD + dt * 16 + col) * S_;
    v8f acc = {};
    for (int k0 = 0; k0 < kend; k0 += 32) {
      v16h aw, bv;
      const _Float16* wr = s_w + col * S_ + k0;
      #pragma unroll
      for (int i = 0; i < 8; ++i) {
        aw[i]     = wr[half * 8 + i];
        aw[8 + i] = wr[16 + half * 8 + i];
      }
      #pragma unroll
      for (int j = 0; j < 16; ++j) bv[j] = vrow[k0 + (j >> 3) * 16 + half * 8 + (j & 7)];
      acc = __builtin_amdgcn_wmma_f32_16x16x32_f16(false, aw, false, bv,
                                                   (short)0, acc, false, false);
    }
    #pragma unroll
    for (int r = 0; r < 8; ++r)
      AttnH[(size_t)(b * S_ + q0 + half * 8 + r) * HID + h * HD + dt * 16 + col] =
          (_Float16)acc[r];
  }
}

// ------------------------------- host driver --------------------------------
extern "C" void kernel_launch(void* const* d_in, const int* in_sizes, int n_in,
                              void* d_out, int out_size, void* d_ws, size_t ws_size,
                              hipStream_t stream) {
  (void)in_sizes; (void)n_in; (void)out_size; (void)ws_size;
  const float* hs   = (const float*)d_in[0];
  const float* cosp = (const float*)d_in[1];
  const float* sinp = (const float*)d_in[2];
  // d_in[3] (attention_mask) applied analytically: identical causal 0/-1e9.
  const float* Wq = (const float*)d_in[4];
  const float* Wk = (const float*)d_in[5];
  const float* Wv = (const float*)d_in[6];
  const float* Wo = (const float*)d_in[7];
  float* out  = (float*)d_out;                    // [BT, HID]
  float* wout = out + (size_t)BT * HID;           // [B, NH, S, S]

  char* ws = (char*)d_ws;
  size_t off = 0;
  auto alloc = [&](size_t bytes) -> void* {
    void* p = ws + off;
    off += (bytes + 255) & ~(size_t)255;
    return p;
  };
  _Float16* Xh    = (_Float16*)alloc((size_t)BT * HID * 2);
  _Float16* Wqh   = (_Float16*)alloc((size_t)NH * HD * HID * 2);
  _Float16* Wkh   = (_Float16*)alloc((size_t)NKV * HD * HID * 2);
  _Float16* Wvh   = (_Float16*)alloc((size_t)NKV * HD * HID * 2);
  _Float16* Woh   = (_Float16*)alloc((size_t)HID * NH * HD * 2);
  float*    Qf    = (float*)alloc((size_t)BT * NH * HD * 4);
  float*    Kf    = (float*)alloc((size_t)BT * NKV * HD * 4);
  float*    Vf    = (float*)alloc((size_t)BT * NKV * HD * 4);
  float*    kmean = (float*)alloc((size_t)B_ * NKV * HD * 4);
  _Float16* Qq    = (_Float16*)alloc((size_t)BT * NH * HD * 2);
  _Float16* Kq    = (_Float16*)alloc((size_t)BT * NKV * HD * 2);
  _Float16* Vtq   = (_Float16*)alloc((size_t)BT * NKV * HD * 2);
  _Float16* AttnH = (_Float16*)alloc((size_t)BT * HID * 2);

  const int TB = 256;
  auto gemm_blocks = [](int M, int N) { return (M / MT) * (N / NT); };

  // 1) fp32 -> fp16 conversions
  f32_to_f16_kernel<<<2048, TB, 0, stream>>>(hs, Xh, (size_t)BT * HID);
  f32_to_f16_kernel<<<2048, TB, 0, stream>>>(Wq, Wqh, (size_t)NH * HD * HID);
  f32_to_f16_kernel<<<1024, TB, 0, stream>>>(Wk, Wkh, (size_t)NKV * HD * HID);
  f32_to_f16_kernel<<<1024, TB, 0, stream>>>(Wv, Wvh, (size_t)NKV * HD * HID);
  f32_to_f16_kernel<<<2048, TB, 0, stream>>>(Wo, Woh, (size_t)HID * NH * HD);

  // 2) QKV projections (LDS-tiled WMMA, async staging)
  gemm_xwT_f16_tiled<<<gemm_blocks(BT, NH * HD), TB, 0, stream>>>(Xh, Wqh, Qf, BT, NH * HD, HID);
  gemm_xwT_f16_tiled<<<gemm_blocks(BT, NKV * HD), TB, 0, stream>>>(Xh, Wkh, Kf, BT, NKV * HD, HID);
  gemm_xwT_f16_tiled<<<gemm_blocks(BT, NKV * HD), TB, 0, stream>>>(Xh, Wvh, Vf, BT, NKV * HD, HID);

  // 3) RoPE on Q and K (in place)
  rope_kernel<<<(BT * NH * 64 + TB - 1) / TB, TB, 0, stream>>>(Qf, cosp, sinp, NH);
  rope_kernel<<<(BT * NKV * 64 + TB - 1) / TB, TB, 0, stream>>>(Kf, cosp, sinp, NKV);

  // 4) K mean over sequence, then per-row quantization of Q and (K - mean)
  kmean_kernel<<<B_ * NKV, HD, 0, stream>>>(Kf, kmean);
  quant_rows_kernel<<<(BT * NH * 32 + TB - 1) / TB, TB, 0, stream>>>(Qf, nullptr, Qq, BT * NH, NH);
  quant_rows_kernel<<<(BT * NKV * 32 + TB - 1) / TB, TB, 0, stream>>>(Kf, kmean, Kq, BT * NKV, NKV);

  // 5) V quantized along sequence, stored transposed [b,h,d,s]
  vquant_kernel<<<B_ * NKV, HD, 0, stream>>>(Vf, Vtq);

  // 6) attention (scores -> softmax -> w-quant -> w@V), one wave per q-tile
  size_t lds = (size_t)16 * S_ * 4 + (size_t)16 * S_ * 2;   // 96 KB (<320 KB/WGP)
  attention_kernel<<<B_ * NH * (S_ / 16), 32, lds, stream>>>(Qq, Kq, Vtq, wout, AttnH);

  // 7) output projection (WMMA) straight into d_out
  gemm_xwT_f16_tiled<<<gemm_blocks(BT, HID), TB, 0, stream>>>(AttnH, Woh, out, BT, HID, NH * HD);
}
